// SelfAttention_21947282882914
// MI455X (gfx1250) — matrix operand order repaired and checked by reading, not compiled, not run on hardware
//
#include <hip/hip_runtime.h>
#include <hip/hip_bf16.h>

// ---------------------------------------------------------------------------
// Problem constants (match reference): B=4, C=2048, D_IN=D_H=D_OUT=1024
// ---------------------------------------------------------------------------
#define BB   4
#define CC   2048
#define DH   1024
#define MQ   (BB * CC)          // 8192 rows for QKV / output projection

typedef __attribute__((ext_vector_type(16))) __bf16   v16bf;
typedef __attribute__((ext_vector_type(8)))  __bf16   v8bf;
typedef __attribute__((ext_vector_type(8)))  float    v8f;
typedef __attribute__((ext_vector_type(4)))  unsigned v4u;
typedef __attribute__((ext_vector_type(8)))  int      v8i;
typedef __attribute__((ext_vector_type(4)))  int      v4i;

#define LDSS 40                 // LDS row stride in bf16 elements (80B, 16B aligned)

// ---- async copy: 32 contiguous bytes global -> LDS (2x b128, ASYNCcnt) ----
__device__ __forceinline__ void async_cp32(const void* gsrc, void* ldst) {
    unsigned long long ga = (unsigned long long)(uintptr_t)gsrc;
    unsigned           la = (unsigned)(uintptr_t)ldst;   // low 32 bits = LDS offset
    asm volatile("global_load_async_to_lds_b128 %0, %1, off\n\t"
                 "global_load_async_to_lds_b128 %0, %1, off offset:16"
                 :
                 : "v"(la), "v"(ga)
                 : "memory");
}
__device__ __forceinline__ void wait_async() {
    asm volatile("s_wait_asynccnt 0x0" ::: "memory");
}

// ---- TDM: 2-D tile DMA, global -> LDS, bf16 elements ----------------------
// Tile: rows x 32 elements (64B rows); LDS padded +16B per row => LDSS stride.
// D# packing per CDNA5 ISA.
// clang-23 toolchain: 6-arg builtin (g0, g1, g2, g3, g_extra, cpol).
__device__ __forceinline__ void tdm_load_tile(const void* gsrc, void* ldst,
                                              unsigned rows, unsigned row_elems,
                                              unsigned tensor_d0, unsigned tensor_d1,
                                              unsigned stride_elems) {
    unsigned long long ga = (unsigned long long)(uintptr_t)gsrc;
    unsigned           la = (unsigned)(uintptr_t)ldst;
    v4u g0;
    g0[0] = 1u;                                            // count=1, no gather
    g0[1] = la;                                            // lds_addr
    g0[2] = (unsigned)ga;                                  // global_addr[31:0]
    g0[3] = (unsigned)((ga >> 32) & 0x01ffffffu) | (2u << 30);  // addr[56:32]|type=2
    v8i g1;
    //  data_size=1(2B) | pad_enable | pad_interval=3(16 dw) | pad_amount=3(4 dw)
    g1[0] = (int)((1u << 16) | (1u << 20) | (3u << 22) | (3u << 25));
    g1[1] = (int)((tensor_d0 & 0xffffu) << 16);            // tensor_dim0[15:0]
    g1[2] = (int)((tensor_d0 >> 16) | ((tensor_d1 & 0xffffu) << 16));
    g1[3] = (int)((tensor_d1 >> 16) | (row_elems << 16));  // tile_dim0
    g1[4] = (int)rows;                                     // tile_dim1 (dim2=0)
    g1[5] = (int)stride_elems;                             // tensor_dim0_stride lo
    g1[6] = 0;
    g1[7] = 0;
    const v4i z4 = {0, 0, 0, 0};
    const v8i z8 = {0, 0, 0, 0, 0, 0, 0, 0};
    __builtin_amdgcn_tensor_load_to_lds(g0, g1, z4, z4, z8, 0);
}
__device__ __forceinline__ void wait_tensor() {
    __builtin_amdgcn_s_wait_tensorcnt(0);
}

// ---- 16-element loader: fp32 global -> bf16 LDS/regs ----------------------
__device__ __forceinline__ void load16(const float* __restrict__ src, __bf16* dst) {
    const float4* p = reinterpret_cast<const float4*>(src);
#pragma unroll
    for (int i = 0; i < 4; ++i) {
        float4 f = p[i];
        dst[4 * i + 0] = (__bf16)f.x;
        dst[4 * i + 1] = (__bf16)f.y;
        dst[4 * i + 2] = (__bf16)f.z;
        dst[4 * i + 3] = (__bf16)f.w;
    }
}
__device__ __forceinline__ void load16(const __bf16* __restrict__ src, __bf16* dst) {
    const v8bf* p = reinterpret_cast<const v8bf*>(src);
    v8bf a = p[0], b = p[1];
    *reinterpret_cast<v8bf*>(dst)     = a;
    *reinterpret_cast<v8bf*>(dst + 8) = b;
}

// ---- 8-element loaders (for the transpose-scatter B path) -----------------
__device__ __forceinline__ void load8(const float* __restrict__ src, __bf16* dst) {
    const float4* p = reinterpret_cast<const float4*>(src);
#pragma unroll
    for (int i = 0; i < 2; ++i) {
        float4 f = p[i];
        dst[4 * i + 0] = (__bf16)f.x;
        dst[4 * i + 1] = (__bf16)f.y;
        dst[4 * i + 2] = (__bf16)f.z;
        dst[4 * i + 3] = (__bf16)f.w;
    }
}
__device__ __forceinline__ void load8(const __bf16* __restrict__ src, __bf16* dst) {
    *reinterpret_cast<v8bf*>(dst) = *reinterpret_cast<const v8bf*>(src);
}

// ---------------------------------------------------------------------------
// Generic tiled GEMM:  out[M x N] = alpha * A[M x Kd] * B + bias
//   BT == false: B row-major [Kd x N]  (transposed while staging into LDS)
//   BT == true : B row-major [N x Kd]  (already N-major; straight copy)
// Staging engine per operand:
//   fp32 source           -> vector load + cvt_pk_bf16 + ds_store
//   bf16 straight copy    -> GLOBAL_LOAD_ASYNC_TO_LDS_B128 (ASYNCcnt)
//   bf16 copy, both sides -> TENSOR_LOAD_TO_LDS, 1 wave issues DMA (TENSORcnt)
// Double-buffered LDS: stage tile t+1 while WMMAs consume tile t.
// Workgroup: 256 threads = 8 wave32; tile 128x128, k-step 32.
// ---------------------------------------------------------------------------
template <typename TA, typename TB, bool BT, bool OUT_BF16, bool HAS_BIAS>
__device__ __forceinline__ void gemm_tile(const TA* __restrict__ A, int lda,
                                          const TB* __restrict__ B, int ldb,
                                          const float* __restrict__ bias,
                                          void* __restrict__ out, int ldo,
                                          float alpha, int Kd, int m0, int n0) {
    __shared__ __bf16 As[2][128 * LDSS];   // [row][k]  (row-major, padded)
    __shared__ __bf16 Bs[2][128 * LDSS];   // [n][k]    (N-major, padded)

    constexpr bool A_ASYNC   = (sizeof(TA) == 2);
    constexpr bool B_ASYNC   = BT && (sizeof(TB) == 2);
    constexpr bool USE_TDM   = A_ASYNC && B_ASYNC;      // scores path
    constexpr bool USE_ASYNC = (A_ASYNC || B_ASYNC) && !USE_TDM;

    const int tid  = threadIdx.x;
    const int lane = tid & 31;
    const int wid  = tid >> 5;
    const int wm   = wid & 3;           // 0..3  -> 32-row strip
    const int wn   = wid >> 2;          // 0..1  -> 64-col strip
    const int lr   = lane & 15;
    const int hh   = lane >> 4;         // half-wave selector

    const v8f vzero = {0.f, 0.f, 0.f, 0.f, 0.f, 0.f, 0.f, 0.f};
    v8f acc[2][4];
#pragma unroll
    for (int mt = 0; mt < 2; ++mt)
#pragma unroll
        for (int nt = 0; nt < 4; ++nt) acc[mt][nt] = vzero;

    // cooperative-load coordinates (fixed per thread)
    const int arow = tid >> 1;                  // 0..127
    const int acol = (tid & 1) * 16;            // 0 or 16
    const int bkp  = (tid >> 4) << 1;           // 0,2,..,30 (BT==false, k-pair)
    const int bno  = (tid & 15) * 8;            // 0..120    (BT==false)

    // -------- one k-tile staging step --------------------------------------
    auto stage = [&](int k0, __bf16* Abuf, __bf16* Bbuf) {
        if constexpr (USE_TDM) {
            if (wid == 0) {
                tdm_load_tile(A + (size_t)m0 * lda + k0, Abuf, 128, 32,
                              (unsigned)lda, 128u, (unsigned)lda);
                tdm_load_tile(B + (size_t)n0 * ldb + k0, Bbuf, 128, 32,
                              (unsigned)ldb, 128u, (unsigned)ldb);
            }
        } else {
            const TA* asrc = A + (size_t)(m0 + arow) * lda + (k0 + acol);
            if constexpr (A_ASYNC) {
                async_cp32(asrc, &Abuf[arow * LDSS + acol]);
            } else {
                load16(asrc, &Abuf[arow * LDSS + acol]);
            }
            if constexpr (BT) {
                const TB* bsrc = B + (size_t)(n0 + arow) * ldb + (k0 + acol);
                if constexpr (B_ASYNC) {
                    async_cp32(bsrc, &Bbuf[arow * LDSS + acol]);
                } else {
                    load16(bsrc, &Bbuf[arow * LDSS + acol]);
                }
            } else {
                // 2(k) x 8(n) patch per thread; pack k-pairs into b32 stores
                __attribute__((aligned(16))) __bf16 r0[8], r1[8];
                load8(B + (size_t)(k0 + bkp) * ldb + (n0 + bno), r0);
                load8(B + (size_t)(k0 + bkp + 1) * ldb + (n0 + bno), r1);
#pragma unroll
                for (int j = 0; j < 8; ++j) {
                    union { __bf16 h[2]; unsigned u; } pk;
                    pk.h[0] = r0[j];
                    pk.h[1] = r1[j];
                    *reinterpret_cast<unsigned*>(&Bbuf[(bno + j) * LDSS + bkp]) = pk.u;
                }
            }
        }
    };
    auto drain = [&]() {
        if constexpr (USE_TDM) {
            if (wid == 0) wait_tensor();
        } else if constexpr (USE_ASYNC) {
            wait_async();
        }
    };

    // -------- prologue: stage first tile -----------------------------------
    stage(0, As[0], Bs[0]);
    drain();
    __syncthreads();

    int cur = 0;
    for (int k0 = 0; k0 < Kd; k0 += 32) {
        const int nxt = cur ^ 1;
        if (k0 + 32 < Kd) stage(k0 + 32, As[nxt], Bs[nxt]);

        // ---- fragments from current buffers -------------------------------
        const __bf16* Ac = As[cur];
        const __bf16* Bc = Bs[cur];
        union { v16bf v; v8bf h[2]; } af[2], bf[4];
#pragma unroll
        for (int mt = 0; mt < 2; ++mt) {
            const __bf16* pa = &Ac[(wm * 32 + mt * 16 + lr) * LDSS + hh * 8];
            af[mt].h[0] = *reinterpret_cast<const v8bf*>(pa);
            af[mt].h[1] = *reinterpret_cast<const v8bf*>(pa + 16);
        }
#pragma unroll
        for (int nt = 0; nt < 4; ++nt) {
            const __bf16* pb = &Bc[(wn * 64 + nt * 16 + lr) * LDSS + hh * 8];
            bf[nt].h[0] = *reinterpret_cast<const v8bf*>(pb);
            bf[nt].h[1] = *reinterpret_cast<const v8bf*>(pb + 16);
        }

        // ---- 8 WMMAs -------------------------------------------------------
#pragma unroll
        for (int mt = 0; mt < 2; ++mt)
#pragma unroll
            for (int nt = 0; nt < 4; ++nt)
                acc[mt][nt] = __builtin_amdgcn_wmma_f32_16x16x32_bf16(
                    false, af[mt].v, false, bf[nt].v,
                    (short)0, acc[mt][nt], false, false);

        drain();            // next tile fully resident in LDS
        __syncthreads();
        cur = nxt;
    }

    // ---- epilogue ---------------------------------------------------------
#pragma unroll
    for (int mt = 0; mt < 2; ++mt) {
#pragma unroll
        for (int nt = 0; nt < 4; ++nt) {
            const int col = n0 + wn * 64 + nt * 16 + lr;
            const float bv = HAS_BIAS ? bias[col] : 0.f;
#pragma unroll
            for (int v = 0; v < 8; ++v) {
                const int row = m0 + wm * 32 + mt * 16 + v + 8 * hh;
                const float val = alpha * acc[mt][nt][v] + bv;
                if (OUT_BF16)
                    reinterpret_cast<__bf16*>(out)[(size_t)row * ldo + col] = (__bf16)val;
                else
                    reinterpret_cast<float*>(out)[(size_t)row * ldo + col] = val;
            }
        }
    }
}

// ---------------------------------------------------------------------------
// Kernel wrappers
// ---------------------------------------------------------------------------
__global__ void k_qkv(const float* __restrict__ x,
                      const float* __restrict__ Wq, const float* __restrict__ bq,
                      const float* __restrict__ Wk, const float* __restrict__ bk,
                      const float* __restrict__ Wv, const float* __restrict__ bv,
                      __bf16* Q, __bf16* K, __bf16* V) {
    const int z = blockIdx.z;
    const float* W   = (z == 0) ? Wq : (z == 1) ? Wk : Wv;
    const float* bia = (z == 0) ? bq : (z == 1) ? bk : bv;
    __bf16*      O   = (z == 0) ? Q  : (z == 1) ? K  : V;
    gemm_tile<float, float, false, true, true>(
        x, DH, W, DH, bia, O, DH, 1.0f, DH,
        blockIdx.y * 128, blockIdx.x * 128);
}

__global__ void k_scores(const __bf16* __restrict__ Q, const __bf16* __restrict__ K,
                         float* S) {
    const int b = blockIdx.z;
    gemm_tile<__bf16, __bf16, true, false, false>(
        Q + (size_t)b * CC * DH, DH, K + (size_t)b * CC * DH, DH, nullptr,
        S + (size_t)b * CC * CC, CC, 0.03125f /* 1/sqrt(1024) */, DH,
        blockIdx.y * 128, blockIdx.x * 128);
}

__global__ void k_softmax(const float* __restrict__ S, __bf16* P, int n) {
    __shared__ float red[256];
    const size_t base = (size_t)blockIdx.x * n;
    const int tid = threadIdx.x;
    float m = -3.402823466e38f;
    for (int c = tid; c < n; c += 256) m = fmaxf(m, S[base + c]);
    red[tid] = m;
    __syncthreads();
    for (int s = 128; s > 0; s >>= 1) {
        if (tid < s) red[tid] = fmaxf(red[tid], red[tid + s]);
        __syncthreads();
    }
    const float rowmax = red[0];
    __syncthreads();
    float sum = 0.f;
    for (int c = tid; c < n; c += 256) sum += __expf(S[base + c] - rowmax);
    red[tid] = sum;
    __syncthreads();
    for (int s = 128; s > 0; s >>= 1) {
        if (tid < s) red[tid] += red[tid + s];
        __syncthreads();
    }
    const float inv = 1.0f / red[0];
    for (int c = tid; c < n; c += 256)
        P[base + c] = (__bf16)(__expf(S[base + c] - rowmax) * inv);
}

__global__ void k_pv(const __bf16* __restrict__ P, const __bf16* __restrict__ V,
                     __bf16* A2) {
    const int b = blockIdx.z;
    gemm_tile<__bf16, __bf16, false, true, false>(
        P + (size_t)b * CC * CC, CC, V + (size_t)b * CC * DH, DH, nullptr,
        A2 + (size_t)b * CC * DH, DH, 1.0f, CC,
        blockIdx.y * 128, blockIdx.x * 128);
}

__global__ void k_out(const __bf16* __restrict__ A2, const float* __restrict__ Wo,
                      const float* __restrict__ bo, float* out) {
    gemm_tile<__bf16, float, false, false, true>(
        A2, DH, Wo, DH, bo, out, DH, 1.0f, DH,
        blockIdx.y * 128, blockIdx.x * 128);
}

// ---------------------------------------------------------------------------
// Host-side launcher
// ---------------------------------------------------------------------------
extern "C" void kernel_launch(void* const* d_in, const int* in_sizes, int n_in,
                              void* d_out, int out_size, void* d_ws, size_t ws_size,
                              hipStream_t stream) {
    const float* x  = (const float*)d_in[0];
    const float* Wq = (const float*)d_in[1];
    const float* bq = (const float*)d_in[2];
    const float* Wk = (const float*)d_in[3];
    const float* bk = (const float*)d_in[4];
    const float* Wv = (const float*)d_in[5];
    const float* bv = (const float*)d_in[6];
    const float* Wo = (const float*)d_in[7];
    const float* bo = (const float*)d_in[8];
    float* out = (float*)d_out;

    // workspace layout
    char* ws = (char*)d_ws;
    const size_t qkv_bytes = (size_t)MQ * DH * sizeof(__bf16);     // 16 MB each
    __bf16* Q  = (__bf16*)(ws);
    __bf16* K  = (__bf16*)(ws + qkv_bytes);
    __bf16* V  = (__bf16*)(ws + 2 * qkv_bytes);
    float*  S  = (float*)(ws + 3 * qkv_bytes);                     // 64 MB
    __bf16* P  = (__bf16*)(ws + 3 * qkv_bytes + (size_t)BB * CC * CC * sizeof(float));
    __bf16* A2 = (__bf16*)((char*)P + (size_t)BB * CC * CC * sizeof(__bf16));

    const dim3 blk(256, 1, 1);

    // 1) Q,K,V = x*W + b   (M=8192, N=1024, K=1024; z selects q/k/v)
    k_qkv<<<dim3(DH / 128, MQ / 128, 3), blk, 0, stream>>>(x, Wq, bq, Wk, bk, Wv, bv,
                                                           Q, K, V);
    // 2) S = Q*K^T / 32    (per-batch 2048x2048x1024, TDM-staged tiles)
    k_scores<<<dim3(CC / 128, CC / 128, BB), blk, 0, stream>>>(Q, K, S);
    // 3) P = softmax(S) rows
    k_softmax<<<dim3(BB * CC, 1, 1), blk, 0, stream>>>(S, P, CC);
    // 4) attn = P*V        (per-batch 2048x1024x2048)
    k_pv<<<dim3(DH / 128, CC / 128, BB), blk, 0, stream>>>(P, V, A2);
    // 5) out = attn*Wo + bo (8192x1024x1024, fp32 out)
    k_out<<<dim3(DH / 128, MQ / 128, 1), blk, 0, stream>>>(A2, Wo, bo, out);
}
